// SmoothLoss_12300786336339
// MI455X (gfx1250) — compile-verified, matrix operand order checked by reading, not run on hardware
//
#include <hip/hip_runtime.h>

#define HH 1536
#define WW 2048
#define TILES_X (WW / 32)            // 64
#define TILES_Y (HH / 32)            // 48
#define NBLK (TILES_X * TILES_Y)     // 3072

typedef __attribute__((ext_vector_type(16))) _Float16 v16h;
typedef __attribute__((ext_vector_type(8)))  float    v8f;

// 12 unique neighbor offsets (each appears twice in the reference PAIRS list)
static constexpr int DYv[12] = { 1, 0, 1,  1, 2, 0, 2,  2, 1,  1, 2,  2 };
static constexpr int DXv[12] = { 0, 1, 1, -1, 0, 2, 1, -1, 2, -2, 2, -2 };

// ---------------------------------------------------------------------------
// Kernel 1: per-tile accumulation (32x32 pixels/block, 2-px halo), then a
// WMMA-based block reduction of the 24 per-thread partial sums:
//   P (32 x 256, rows 24..31 zero) x ones(256)  via v_wmma_f32_16x16x32_f16,
//   with hi/lo f16 split of each f32 operand for near-f32 accuracy.
// Per-block partials land in ws[row * NBLK + blockIdx.x].
// ---------------------------------------------------------------------------
__global__ __launch_bounds__(256)
void smooth_tile_kernel(const float* __restrict__ inp,
                        const float* __restrict__ outp,
                        float* __restrict__ ws)
{
    __shared__ float sY [34 * 36];
    __shared__ float sCb[34 * 36];
    __shared__ float sCr[34 * 36];
    __shared__ float sR [34 * 36];
    __shared__ float sG [34 * 36];
    __shared__ float sB [34 * 36];
    __shared__ float P[32][256];

    const int tid = threadIdx.x;
    const int bx  = blockIdx.x % TILES_X;
    const int by  = blockIdx.x / TILES_X;
    const int gy0 = by * 32;          // tile row 0
    const int gx0 = bx * 32 - 2;      // LDS column 0 maps to gx0

    const float* inR = inp;
    const float* inG = inp + HH * WW;
    const float* inB = inp + 2 * HH * WW;
    const float* oR  = outp;
    const float* oG  = outp + HH * WW;
    const float* oB  = outp + 2 * HH * WW;

    // ---- load (34 x 36) halo region, converting RGB -> YCbCr on the fly ----
    for (int idx = tid; idx < 34 * 36; idx += 256) {
        const int ly = idx / 36;
        const int lx = idx - ly * 36;
        const int gy = gy0 + ly;
        const int gx = gx0 + lx;
        float r = 0.f, g = 0.f, b = 0.f, pr = 0.f, pg = 0.f, pb = 0.f;
        if (gy < HH && (unsigned)gx < (unsigned)WW) {
            const int o = gy * WW + gx;
            r = inR[o]; g = inG[o]; b = inB[o];
            pr = oR[o]; pg = oG[o]; pb = oB[o];
        }
        sY [idx] =  0.257f * r + 0.564f * g + 0.098f * b + (16.0f  / 255.0f);
        sCb[idx] = -0.148f * r - 0.291f * g + 0.439f * b + (128.0f / 255.0f);
        sCr[idx] =  0.439f * r - 0.368f * g - 0.071f * b + (128.0f / 255.0f);
        sR [idx] = pr;
        sG [idx] = pg;
        sB [idx] = pb;
    }
    __syncthreads();

    // ---- per-thread accumulation: 4 pixels x 12 offsets -------------------
    float accW[12], accA[12];
#pragma unroll
    for (int k = 0; k < 12; ++k) { accW[k] = 0.f; accA[k] = 0.f; }

    const int tx  = tid & 31;
    const int ty0 = tid >> 5;
#pragma unroll
    for (int i = 0; i < 4; ++i) {
        const int ty = ty0 + 8 * i;
        const int gy = gy0 + ty;
        const int gx = bx * 32 + tx;
        const int c0 = ty * 36 + (tx + 2);
        const float y0  = sY[c0],  cb0 = sCb[c0], cr0 = sCr[c0];
        const float r0  = sR[c0],  g0  = sG[c0],  b0  = sB[c0];
#pragma unroll
        for (int k = 0; k < 12; ++k) {
            const int dy = DYv[k], dx = DXv[k];
            const int n  = c0 + dy * 36 + dx;
            const float d0 = y0  - sY[n];
            const float d1 = cb0 - sCb[n];
            const float d2 = cr0 - sCr[n];
            const float ssd = d0 * d0 + d1 * d1 + d2 * d2;
            const float w   = __expf(ssd * -0.005f);           // -0.5*sigma^2
            const float a   = fabsf(r0 - sR[n]) + fabsf(g0 - sG[n])
                            + fabsf(b0 - sB[n]);
            const bool valid = (gy + dy < HH) &&
                               ((unsigned)(gx + dx) < (unsigned)WW);
            if (valid) { accW[k] += w; accA[k] += a; }
        }
    }

    // ---- stage partials: P[row][thread], rows 24..31 zero-padded ----------
#pragma unroll
    for (int k = 0; k < 12; ++k) {
        P[k][tid]      = accW[k];
        P[12 + k][tid] = accA[k];
    }
#pragma unroll
    for (int k = 24; k < 32; ++k) P[k][tid] = 0.f;
    __syncthreads();

    // ---- wave 0: row sums of P via WMMA (D = A x ones + C) ----------------
    if (tid < 32) {
        const int lane = tid;
        const int Mrow = lane & 15;
        const int kb   = (lane < 16) ? 0 : 8;   // K-base per A-fragment layout

        v16h ones;
#pragma unroll
        for (int h = 0; h < 16; ++h) ones[h] = (_Float16)1.0f;

#pragma unroll
        for (int g2 = 0; g2 < 2; ++g2) {        // row groups 0..15 / 16..31
            v8f acc = {0.f, 0.f, 0.f, 0.f, 0.f, 0.f, 0.f, 0.f};
#pragma unroll
            for (int ch = 0; ch < 8; ++ch) {    // K chunks of 32 columns
                v16h ahi, alo;
#pragma unroll
                for (int h = 0; h < 16; ++h) {
                    // 16-bit A 16x32 layout: K = kb + (h<8 ? h : h+8)
                    const int col = ch * 32 + kb + ((h < 8) ? h : h + 8);
                    const float x = P[g2 * 16 + Mrow][col];
                    const _Float16 hi = (_Float16)x;
                    ahi[h] = hi;
                    alo[h] = (_Float16)(x - (float)hi);
                }
                acc = __builtin_amdgcn_wmma_f32_16x16x32_f16(
                        false, ahi, false, ones, (short)0, acc, false, false);
                acc = __builtin_amdgcn_wmma_f32_16x16x32_f16(
                        false, alo, false, ones, (short)0, acc, false, false);
            }
            // D layout: VGPR r -> lane 0 holds rows M=0..7, lane 16 rows 8..15
            if (lane == 0 || lane == 16) {
                const int base = g2 * 16 + ((lane == 0) ? 0 : 8);
#pragma unroll
                for (int r = 0; r < 8; ++r) {
                    const int row = base + r;
                    if (row < 24) ws[row * NBLK + blockIdx.x] = acc[r];
                }
            }
        }
    }
}

// ---------------------------------------------------------------------------
// Kernel 2: deterministic tree reduction over the 3072 block partials and
// final scaling: total = sum_k 2 * sumW_k * sumA_k / (3 * N_k^2)
// ---------------------------------------------------------------------------
__global__ __launch_bounds__(256)
void smooth_final_kernel(const float* __restrict__ ws, float* __restrict__ res)
{
    __shared__ float red[256];
    const int tid = threadIdx.x;
    float total = 0.f;

    for (int k = 0; k < 12; ++k) {
        float pw = 0.f, pa = 0.f;
        for (int b = tid; b < NBLK; b += 256) {
            pw += ws[k * NBLK + b];
            pa += ws[(12 + k) * NBLK + b];
        }
        red[tid] = pw;
        __syncthreads();
        for (int s = 128; s > 0; s >>= 1) {
            if (tid < s) red[tid] += red[tid + s];
            __syncthreads();
        }
        const float sw = red[0];
        __syncthreads();
        red[tid] = pa;
        __syncthreads();
        for (int s = 128; s > 0; s >>= 1) {
            if (tid < s) red[tid] += red[tid + s];
            __syncthreads();
        }
        const float sa = red[0];
        __syncthreads();

        if (tid == 0) {
            const int adx = (DXv[k] < 0) ? -DXv[k] : DXv[k];
            const double n = (double)(HH - DYv[k]) * (double)(WW - adx);
            total += (float)(2.0 * (double)sw * (double)sa / (3.0 * n * n));
        }
    }
    if (tid == 0) res[0] = total;
}

extern "C" void kernel_launch(void* const* d_in, const int* in_sizes, int n_in,
                              void* d_out, int out_size, void* d_ws, size_t ws_size,
                              hipStream_t stream) {
    (void)in_sizes; (void)n_in; (void)out_size; (void)ws_size;
    const float* inp  = (const float*)d_in[0];
    const float* outp = (const float*)d_in[1];
    float* ws  = (float*)d_ws;   // needs 24 * NBLK * 4 = 294,912 bytes
    float* res = (float*)d_out;

    smooth_tile_kernel <<<NBLK, 256, 0, stream>>>(inp, outp, ws);
    smooth_final_kernel<<<1,    256, 0, stream>>>(ws, res);
}
